// EquivariantTransformerBackBone_62380105007999
// MI455X (gfx1250) — matrix operand order repaired and checked by reading, not compiled, not run on hardware
//
#include <hip/hip_runtime.h>
#include <hip/hip_bf16.h>
#include <math.h>

// ---------------------------------------------------------------------------
// EquivariantTransformerBackBone for MI455X (gfx1250, wave32, WMMA).
// All dense algebra goes through v_wmma_f32_16x16x32_bf16 (bf16 in, f32 acc).
// GEMM: 4 independent accumulator chains per wave (64-row M stripe) and
// explicit register double-buffering so next-k loads overlap current WMMAs.
// ---------------------------------------------------------------------------

typedef __bf16 bf16_t;
typedef __attribute__((ext_vector_type(16))) __bf16 v16bf;
typedef __attribute__((ext_vector_type(8)))  __bf16 v8bf;
typedef __attribute__((ext_vector_type(8)))  float  v8f;

#define N_ATOMS 16384
#define N_EDGES 131072
#define HDIM    128
#define NRBF    64
#define NLAYERS 4

__device__ __forceinline__ float silu_f(float x) { return x / (1.0f + __expf(-x)); }

// ---------------------------------------------------------------------------
// Pack a row-major f32 weight (K x N) into WMMA B-fragment order (bf16).
// Per ISA 7.12.2 (16-bit B 32x16): lane -> N = lane%16; element j ->
// K = 8*half + j (j<8), K = 16 + 8*half + (j-8) (j>=8).
// Layout: Bp[((nt*KT + kt)*32 + lane)*16 + j]
// ---------------------------------------------------------------------------
__global__ void pack_b(const float* __restrict__ B, bf16_t* __restrict__ Bp,
                       int K, int N) {
  int tid = blockIdx.x * blockDim.x + threadIdx.x;
  int KT = K >> 5, NT = N >> 4;
  if (tid >= KT * NT * 32) return;
  int lane = tid & 31, tile = tid >> 5;
  int kt = tile % KT, nt = tile / KT;
  int n  = (nt << 4) + (lane & 15);
  int kb = (kt << 5) + ((lane >> 4) << 3);
  bf16_t* d = Bp + (size_t)tid * 16;
#pragma unroll
  for (int j = 0; j < 16; ++j) d[j] = (bf16_t)B[(size_t)(kb + (j & 7) + ((j >> 3) << 4)) * N + n];
}

// ---------------------------------------------------------------------------
// WMMA GEMM: C[M,N] = act(A[M,K] @ B[K,N] + bias).  A row-major bf16,
// B pre-packed bf16 (pack_b), C f32.  M%64==0, N%128==0, K%32==0.
// Block = (32,8): 8 waves; block tile = 64(M) x 128(N); each wave owns one
// 16-col tile and a 64-row stripe = 4 independent 16x16 accumulators.
// Register double-buffered: k-step i+1 fragment loads are issued before the
// k-step i WMMAs so loads overlap matrix ops (partial s_wait_loadcnt).
// A fragment per ISA 7.12.2: lane half h holds K = 8h..8h+7 and 16+8h..16+8h+7.
// ---------------------------------------------------------------------------
struct AFrag { v8bf h[2]; };

__device__ __forceinline__ void load_afrag(AFrag& f, const bf16_t* p) {
  f.h[0] = *(const v8bf*)(p);
  f.h[1] = *(const v8bf*)(p + 16);
}
__device__ __forceinline__ v16bf afrag_v(const AFrag& f) {
  union { AFrag a; v16bf v; } u;
  u.a = f;
  return u.v;
}

__global__ __launch_bounds__(256)
void gemm_wmma(const bf16_t* __restrict__ A, const bf16_t* __restrict__ Bp,
               const float* __restrict__ bias, float* __restrict__ Cmat,
               int M, int N, int K, int act) {
  const int lane = threadIdx.x;
  const int wave = threadIdx.y;
  const int row0 = blockIdx.x << 6;         // 64-row stripe
  const int ct   = blockIdx.y * 8 + wave;   // global 16-col tile index
  const int KT   = K >> 5;
  const int m    = lane & 15;
  const int half = lane >> 4;

  const bf16_t* ap = A + (size_t)(row0 + m) * K + half * 8;
  const bf16_t* bp = Bp + ((size_t)ct * KT * 32 + lane) * 16;
  const size_t  arow16 = (size_t)16 * K;    // 16 rows down

  v8f acc0 = {0.f, 0.f, 0.f, 0.f, 0.f, 0.f, 0.f, 0.f};
  v8f acc1 = acc0, acc2 = acc0, acc3 = acc0;

  // prologue: load k-step 0 fragment set
  v16bf cb = *(const v16bf*)(bp);
  AFrag ca0, ca1, ca2, ca3;
  load_afrag(ca0, ap);
  load_afrag(ca1, ap + arow16);
  load_afrag(ca2, ap + 2 * arow16);
  load_afrag(ca3, ap + 3 * arow16);

  for (int kt = 0; kt < KT; ++kt) {
    // issue next k-step loads first (dummy re-load of current on last iter,
    // uniform pointer select -> EXEC stays all-ones, always in-bounds)
    const bool more = (kt + 1 < KT);
    const bf16_t* apn = more ? (ap + 32) : ap;
    const bf16_t* bpn = more ? (bp + 32 * 16) : bp;
    v16bf nb = *(const v16bf*)(bpn);
    AFrag na0, na1, na2, na3;
    load_afrag(na0, apn);
    load_afrag(na1, apn + arow16);
    load_afrag(na2, apn + 2 * arow16);
    load_afrag(na3, apn + 3 * arow16);

    // consume current set: 4 independent WMMA chains (XDL pipelining)
    acc0 = __builtin_amdgcn_wmma_f32_16x16x32_bf16(false, afrag_v(ca0), false, cb,
                                                   (short)0, acc0, false, false);
    acc1 = __builtin_amdgcn_wmma_f32_16x16x32_bf16(false, afrag_v(ca1), false, cb,
                                                   (short)0, acc1, false, false);
    acc2 = __builtin_amdgcn_wmma_f32_16x16x32_bf16(false, afrag_v(ca2), false, cb,
                                                   (short)0, acc2, false, false);
    acc3 = __builtin_amdgcn_wmma_f32_16x16x32_bf16(false, afrag_v(ca3), false, cb,
                                                   (short)0, acc3, false, false);

    // rotate buffers (renamed away after unrolling)
    ca0 = na0; ca1 = na1; ca2 = na2; ca3 = na3; cb = nb;
    ap = apn; bp = bpn;
  }

  const int col = (ct << 4) + m;
  const float bias_v = bias ? bias[col] : 0.0f;
  v8f accs[4] = {acc0, acc1, acc2, acc3};
#pragma unroll
  for (int t = 0; t < 4; ++t) {
#pragma unroll
    for (int r = 0; r < 8; ++r) {
      float xo = accs[t][r] + bias_v;
      if (act) xo = silu_f(xo);
      // C layout: VGPR r -> M = r + 8*half (ISA 7.12.2)
      Cmat[(size_t)(row0 + t * 16 + r + 8 * half) * N + col] = xo;
    }
  }
}

// ---------------------------------------------------------------------------
// Edge geometry + cosine cutoff + exp-normal RBF (written directly as bf16).
// ---------------------------------------------------------------------------
__global__ void edge_geom_rbf(const float* __restrict__ pos,
                              const int* __restrict__ snd, const int* __restrict__ rcv,
                              bf16_t* __restrict__ ea, float* __restrict__ Cw,
                              float* __restrict__ evec) {
  int e = blockIdx.x * blockDim.x + threadIdx.x;
  if (e >= N_EDGES) return;
  int s = snd[e], r = rcv[e];
  float vx = pos[r * 3 + 0] - pos[s * 3 + 0];
  float vy = pos[r * 3 + 1] - pos[s * 3 + 1];
  float vz = pos[r * 3 + 2] - pos[s * 3 + 2];
  const float eps = 1e-6f;
  float ax = vx + eps, ay = vy + eps, az = vz + eps;
  float w = sqrtf(ax * ax + ay * ay + az * az);          // edge_weight
  float C = (w < 10.0f) ? 0.5f * (__cosf(w * 0.31415926535f) + 1.0f) : 0.0f;
  Cw[e] = C;
  float mew = (s == r) ? 1.0f : w;
  mew = fmaxf(mew, 0.01f);
  evec[e * 3 + 0] = vx / mew;
  evec[e * 3 + 1] = vy / mew;
  evec[e * 3 + 2] = vz / mew;
  const float start = 4.5399929762484854e-05f;           // exp(-10)
  const float step  = (1.0f - start) * (1.0f / 63.0f);
  const float beta  = 1024.09296f;                       // ((2/64)*(1-start))^-2
  float ew = __expf(-0.5f * w);                          // exp(-alpha*d), alpha=0.5
  bf16_t* row = ea + (size_t)e * NRBF;
  for (int j = 0; j < NRBF; ++j) {
    float d = ew - (start + j * step);
    row[j] = (bf16_t)(C * __expf(-beta * d * d));
  }
}

// msg = Wmsg * C * embed2[z[send]]; atomic scatter into x_nb[recv]
__global__ void nb_msg_scatter(const float* __restrict__ wmsg, const float* __restrict__ Cw,
                               const int* __restrict__ zn, const float* __restrict__ emb2,
                               const int* __restrict__ snd, const int* __restrict__ rcv,
                               float* __restrict__ xnb) {
  size_t i = (size_t)blockIdx.x * blockDim.x + threadIdx.x;
  if (i >= (size_t)N_EDGES * HDIM) return;
  int e = (int)(i >> 7), h = (int)(i & 127);
  float m = wmsg[i] * Cw[e] * emb2[(size_t)zn[snd[e]] * HDIM + h];
  atomicAdd(&xnb[(size_t)rcv[e] * HDIM + h], m);
}

// xcat = [embed1[z] | x_nb] as bf16 (N, 256)
__global__ void build_xcat(const int* __restrict__ zn, const float* __restrict__ emb1,
                           const float* __restrict__ xnb, bf16_t* __restrict__ xcat) {
  int i = blockIdx.x * blockDim.x + threadIdx.x;
  if (i >= N_ATOMS * HDIM) return;
  int nr = i >> 7, h = i & 127;
  xcat[(size_t)nr * 256 + h]       = (bf16_t)emb1[(size_t)zn[nr] * HDIM + h];
  xcat[(size_t)nr * 256 + 128 + h] = (bf16_t)xnb[i];
}

// Row layernorm over H=128. One wave per row (4 values/lane). OUT_BF: bf16 out.
template <int OUT_BF>
__global__ void layernorm_k(const float* __restrict__ x, const float* __restrict__ g,
                            const float* __restrict__ b, void* __restrict__ out) {
  int row  = blockIdx.x * 8 + (threadIdx.x >> 5);
  int lane = threadIdx.x & 31;
  const float* xr = x + (size_t)row * HDIM;
  float v0 = xr[lane], v1 = xr[lane + 32], v2 = xr[lane + 64], v3 = xr[lane + 96];
  float s = v0 + v1 + v2 + v3;
  for (int o = 16; o; o >>= 1) s += __shfl_xor(s, o, 32);
  float mean = s * (1.0f / HDIM);
  float d0 = v0 - mean, d1 = v1 - mean, d2 = v2 - mean, d3 = v3 - mean;
  float vs = d0 * d0 + d1 * d1 + d2 * d2 + d3 * d3;
  for (int o = 16; o; o >>= 1) vs += __shfl_xor(vs, o, 32);
  float inv = rsqrtf(vs * (1.0f / HDIM) + 1e-6f);
  float o0 = d0 * inv * g[lane]      + b[lane];
  float o1 = d1 * inv * g[lane + 32] + b[lane + 32];
  float o2 = d2 * inv * g[lane + 64] + b[lane + 64];
  float o3 = d3 * inv * g[lane + 96] + b[lane + 96];
  if (OUT_BF) {
    bf16_t* ob = (bf16_t*)out + (size_t)row * HDIM;
    ob[lane] = (bf16_t)o0; ob[lane + 32] = (bf16_t)o1;
    ob[lane + 64] = (bf16_t)o2; ob[lane + 96] = (bf16_t)o3;
  } else {
    float* of = (float*)out + (size_t)row * HDIM;
    of[lane] = o0; of[lane + 32] = o1; of[lane + 64] = o2; of[lane + 96] = o3;
  }
}

// per-head (16 contiguous channels) L2 normalization, in place
__global__ void l2norm_heads(float* __restrict__ q, int total) {
  int i = blockIdx.x * blockDim.x + threadIdx.x;
  if (i >= total) return;
  float val = q[i];
  float s = val * val;
  s += __shfl_xor(s, 1, 32);
  s += __shfl_xor(s, 2, 32);
  s += __shfl_xor(s, 4, 32);
  s += __shfl_xor(s, 8, 32);
  q[i] = val / (sqrtf(s) + 1e-6f);
}

// CoorsNorm: vn = vec / sqrt(max(sum_s vec^2, 1e-8)) * scale[l], rows (n*3+s)
__global__ void coorsnorm_k(const float* __restrict__ vec, const float* __restrict__ csc,
                            int l, bf16_t* __restrict__ vnbf) {
  int i = blockIdx.x * blockDim.x + threadIdx.x;
  if (i >= N_ATOMS * HDIM) return;
  int nr = i >> 7, h = i & 127;
  float a = vec[((size_t)nr * 3 + 0) * HDIM + h];
  float b = vec[((size_t)nr * 3 + 1) * HDIM + h];
  float c = vec[((size_t)nr * 3 + 2) * HDIM + h];
  float inv = csc[l] * rsqrtf(fmaxf(a * a + b * b + c * c, 1e-8f));
  vnbf[((size_t)nr * 3 + 0) * HDIM + h] = (bf16_t)(a * inv);
  vnbf[((size_t)nr * 3 + 1) * HDIM + h] = (bf16_t)(b * inv);
  vnbf[((size_t)nr * 3 + 2) * HDIM + h] = (bf16_t)(c * inv);
}

// vec_dot[n,h] = sum_s vp[n,s,h] * vp[n,s,128+h]   (vp rows (n*3+s), width 384)
__global__ void vec_dot_k(const float* __restrict__ vp, float* __restrict__ vd) {
  int i = blockIdx.x * blockDim.x + threadIdx.x;
  if (i >= N_ATOMS * HDIM) return;
  int nr = i >> 7, h = i & 127;
  float s = 0.f;
#pragma unroll
  for (int d3 = 0; d3 < 3; ++d3) {
    const float* rowp = vp + ((size_t)nr * 3 + d3) * 384;
    s += rowp[h] * rowp[128 + h];
  }
  vd[i] = s;
}

// Edge attention + message scatter.  Block = one edge, 128 threads (h).
// attn head reduce over 16-lane groups via shfl_xor.
__global__ void edge_attn_scatter(const float* __restrict__ q, const float* __restrict__ k,
                                  const float* __restrict__ v, const float* __restrict__ dk,
                                  const float* __restrict__ dv, const float* __restrict__ vec,
                                  const float* __restrict__ evec, const float* __restrict__ Cw,
                                  const int* __restrict__ snd, const int* __restrict__ rcv,
                                  float* __restrict__ xagg, float* __restrict__ vagg) {
  int e = blockIdx.x;
  int h = threadIdx.x;
  int s = snd[e], r = rcv[e];
  float a = q[(size_t)r * HDIM + h] * k[(size_t)s * HDIM + h] * dk[(size_t)e * HDIM + h];
  a += __shfl_xor(a, 1, 32);
  a += __shfl_xor(a, 2, 32);
  a += __shfl_xor(a, 4, 32);
  a += __shfl_xor(a, 8, 32);
  float attn = silu_f(a) * Cw[e];
  float xh = v[(size_t)s * 384 + h] * dv[(size_t)e * 384 + h] * attn;
  atomicAdd(&xagg[(size_t)r * HDIM + h], xh);
  float s1 = v[(size_t)s * 384 + 128 + h] * dv[(size_t)e * 384 + 128 + h];
  float s2 = v[(size_t)s * 384 + 256 + h] * dv[(size_t)e * 384 + 256 + h];
#pragma unroll
  for (int d3 = 0; d3 < 3; ++d3) {
    float vj = vec[((size_t)s * 3 + d3) * HDIM + h] * s1 + s2 * evec[e * 3 + d3];
    atomicAdd(&vagg[((size_t)r * 3 + d3) * HDIM + h], vj);
  }
}

__global__ void f32_to_bf16_k(const float* __restrict__ s, bf16_t* __restrict__ d, int n) {
  int i = blockIdx.x * blockDim.x + threadIdx.x;
  if (i < n) d[i] = (bf16_t)s[i];
}

// x += vec_dot*o2 + o3;  vec += vec3*o1 + vec_agg   (vec3 = vp[:,:,256:384])
__global__ void layer_update(float* __restrict__ x, float* __restrict__ vec,
                             const float* __restrict__ o, const float* __restrict__ vdot,
                             const float* __restrict__ vp, const float* __restrict__ vagg) {
  int i = blockIdx.x * blockDim.x + threadIdx.x;
  if (i >= N_ATOMS * HDIM) return;
  int nr = i >> 7, h = i & 127;
  float o1 = o[(size_t)nr * 384 + h];
  float o2 = o[(size_t)nr * 384 + 128 + h];
  float o3 = o[(size_t)nr * 384 + 256 + h];
  x[i] += vdot[i] * o2 + o3;
#pragma unroll
  for (int d3 = 0; d3 < 3; ++d3) {
    size_t vi = ((size_t)nr * 3 + d3) * HDIM + h;
    vec[vi] += vp[((size_t)nr * 3 + d3) * 384 + 256 + h] * o1 + vagg[vi];
  }
}

// ---------------------------------------------------------------------------
extern "C" void kernel_launch(void* const* d_in, const int* in_sizes, int n_in,
                              void* d_out, int out_size, void* d_ws, size_t ws_size,
                              hipStream_t stream) {
  (void)in_sizes; (void)n_in; (void)out_size; (void)ws_size;

  const int*   zn    = (const int*)d_in[0];
  const float* pos   = (const float*)d_in[1];
  const int*   snd   = (const int*)d_in[2];
  const int*   rcv   = (const int*)d_in[3];
  const float* emb1  = (const float*)d_in[4];
  const float* emb2  = (const float*)d_in[5];
  const float* Wdist = (const float*)d_in[6];
  const float* bdist = (const float*)d_in[7];
  const float* Wcomb = (const float*)d_in[8];
  const float* bcomb = (const float*)d_in[9];
  const float* ln_g  = (const float*)d_in[10];
  const float* ln_b  = (const float*)d_in[11];
  const float* Wq    = (const float*)d_in[12];
  const float* bq    = (const float*)d_in[13];
  const float* Wk    = (const float*)d_in[14];
  const float* bk    = (const float*)d_in[15];
  const float* Wv    = (const float*)d_in[16];
  const float* bvv   = (const float*)d_in[17];
  const float* Wvec  = (const float*)d_in[18];
  const float* Wdk   = (const float*)d_in[19];
  const float* bdk   = (const float*)d_in[20];
  const float* Wdv   = (const float*)d_in[21];
  const float* bdv   = (const float*)d_in[22];
  const float* Wo    = (const float*)d_in[23];
  const float* bo    = (const float*)d_in[24];
  const float* csc   = (const float*)d_in[25];
  const float* lnfg  = (const float*)d_in[26];
  const float* lnfb  = (const float*)d_in[27];

  char* p = (char*)d_ws;
  auto carve = [&](size_t bytes) -> char* {
    char* r = p;
    p += (bytes + 255) & ~(size_t)255;
    return r;
  };

  bf16_t* ea     = (bf16_t*)carve((size_t)N_EDGES * NRBF * 2);
  float*  Cw     = (float*)carve((size_t)N_EDGES * 4);
  float*  evec   = (float*)carve((size_t)N_EDGES * 3 * 4);
  float*  xnb    = (float*)carve((size_t)N_ATOMS * HDIM * 4);
  bf16_t* xcat   = (bf16_t*)carve((size_t)N_ATOMS * 2 * HDIM * 2);
  float*  x      = (float*)carve((size_t)N_ATOMS * HDIM * 4);
  bf16_t* xnbf   = (bf16_t*)carve((size_t)N_ATOMS * HDIM * 2);
  float*  q      = (float*)carve((size_t)N_ATOMS * HDIM * 4);
  float*  kbuf   = (float*)carve((size_t)N_ATOMS * HDIM * 4);
  float*  v      = (float*)carve((size_t)N_ATOMS * 3 * HDIM * 4);
  bf16_t* vnbf   = (bf16_t*)carve((size_t)N_ATOMS * 3 * HDIM * 2);
  float*  vp     = (float*)carve((size_t)N_ATOMS * 3 * 3 * HDIM * 4);
  float*  vdot   = (float*)carve((size_t)N_ATOMS * HDIM * 4);
  float*  dk     = (float*)carve((size_t)N_EDGES * HDIM * 4);      // also Wmsg
  float*  dv     = (float*)carve((size_t)N_EDGES * 3 * HDIM * 4);
  float*  xagg   = (float*)carve((size_t)N_ATOMS * HDIM * 4);
  bf16_t* xaggbf = (bf16_t*)carve((size_t)N_ATOMS * HDIM * 2);
  float*  vagg   = (float*)carve((size_t)N_ATOMS * 3 * HDIM * 4);
  float*  vec    = (float*)carve((size_t)N_ATOMS * 3 * HDIM * 4);
  float*  obuf   = (float*)carve((size_t)N_ATOMS * 3 * HDIM * 4);

  bf16_t* pk_dist = (bf16_t*)carve((size_t)64 * 128 * 2);
  bf16_t* pk_comb = (bf16_t*)carve((size_t)256 * 128 * 2);
  bf16_t *pkq[NLAYERS], *pkk[NLAYERS], *pkv[NLAYERS], *pkvec[NLAYERS];
  bf16_t *pkdk[NLAYERS], *pkdv[NLAYERS], *pko[NLAYERS];
  for (int l = 0; l < NLAYERS; ++l) {
    pkq[l]   = (bf16_t*)carve((size_t)128 * 128 * 2);
    pkk[l]   = (bf16_t*)carve((size_t)128 * 128 * 2);
    pkv[l]   = (bf16_t*)carve((size_t)128 * 384 * 2);
    pkvec[l] = (bf16_t*)carve((size_t)128 * 384 * 2);
    pkdk[l]  = (bf16_t*)carve((size_t)64 * 128 * 2);
    pkdv[l]  = (bf16_t*)carve((size_t)64 * 384 * 2);
    pko[l]   = (bf16_t*)carve((size_t)128 * 384 * 2);
  }

  auto packW = [&](const float* W, bf16_t* Bp, int K, int N) {
    int total = (K >> 5) * (N >> 4) * 32;
    pack_b<<<(total + 255) / 256, 256, 0, stream>>>(W, Bp, K, N);
  };
  auto gemm = [&](const bf16_t* A, const bf16_t* Bp, const float* bias, float* Cm,
                  int M, int N, int K, int act) {
    dim3 g(M / 64, N / 128), b(32, 8);
    gemm_wmma<<<g, b, 0, stream>>>(A, Bp, bias, Cm, M, N, K, act);
  };

  // --- pack weights (cheap; deterministic every call) ---
  packW(Wdist, pk_dist, 64, 128);
  packW(Wcomb, pk_comb, 256, 128);
  for (int l = 0; l < NLAYERS; ++l) {
    packW(Wq + (size_t)l * 128 * 128, pkq[l], 128, 128);
    packW(Wk + (size_t)l * 128 * 128, pkk[l], 128, 128);
    packW(Wv + (size_t)l * 128 * 384, pkv[l], 128, 384);
    packW(Wvec + (size_t)l * 128 * 384, pkvec[l], 128, 384);
    packW(Wdk + (size_t)l * 64 * 128, pkdk[l], 64, 128);
    packW(Wdv + (size_t)l * 64 * 384, pkdv[l], 64, 384);
    packW(Wo + (size_t)l * 128 * 384, pko[l], 128, 384);
  }

  // --- stage 0: edges + neighbor embedding ---
  hipMemsetAsync(xnb, 0, (size_t)N_ATOMS * HDIM * 4, stream);
  hipMemsetAsync(vec, 0, (size_t)N_ATOMS * 3 * HDIM * 4, stream);

  edge_geom_rbf<<<(N_EDGES + 255) / 256, 256, 0, stream>>>(pos, snd, rcv, ea, Cw, evec);

  gemm(ea, pk_dist, bdist, dk /*Wmsg*/, N_EDGES, 128, 64, 0);
  nb_msg_scatter<<<(int)(((size_t)N_EDGES * HDIM + 255) / 256), 256, 0, stream>>>(
      dk, Cw, zn, emb2, snd, rcv, xnb);
  build_xcat<<<(N_ATOMS * HDIM + 255) / 256, 256, 0, stream>>>(zn, emb1, xnb, xcat);
  gemm(xcat, pk_comb, bcomb, x, N_ATOMS, 128, 256, 0);

  // --- attention layers ---
  for (int l = 0; l < NLAYERS; ++l) {
    layernorm_k<1><<<N_ATOMS / 8, 256, 0, stream>>>(x, ln_g + l * HDIM, ln_b + l * HDIM, xnbf);

    gemm(xnbf, pkq[l], bq + (size_t)l * HDIM, q, N_ATOMS, 128, 128, 0);
    gemm(xnbf, pkk[l], bk + (size_t)l * HDIM, kbuf, N_ATOMS, 128, 128, 0);
    gemm(xnbf, pkv[l], bvv + (size_t)l * 3 * HDIM, v, N_ATOMS, 384, 128, 0);

    l2norm_heads<<<(N_ATOMS * HDIM + 255) / 256, 256, 0, stream>>>(q, N_ATOMS * HDIM);
    l2norm_heads<<<(N_ATOMS * HDIM + 255) / 256, 256, 0, stream>>>(kbuf, N_ATOMS * HDIM);

    coorsnorm_k<<<(N_ATOMS * HDIM + 255) / 256, 256, 0, stream>>>(vec, csc, l, vnbf);
    gemm(vnbf, pkvec[l], nullptr, vp, N_ATOMS * 3, 384, 128, 0);
    vec_dot_k<<<(N_ATOMS * HDIM + 255) / 256, 256, 0, stream>>>(vp, vdot);

    gemm(ea, pkdk[l], bdk + (size_t)l * HDIM, dk, N_EDGES, 128, 64, 1);
    gemm(ea, pkdv[l], bdv + (size_t)l * 3 * HDIM, dv, N_EDGES, 384, 64, 1);

    hipMemsetAsync(xagg, 0, (size_t)N_ATOMS * HDIM * 4, stream);
    hipMemsetAsync(vagg, 0, (size_t)N_ATOMS * 3 * HDIM * 4, stream);
    edge_attn_scatter<<<N_EDGES, 128, 0, stream>>>(q, kbuf, v, dk, dv, vec, evec, Cw,
                                                   snd, rcv, xagg, vagg);

    f32_to_bf16_k<<<(N_ATOMS * HDIM + 255) / 256, 256, 0, stream>>>(xagg, xaggbf,
                                                                    N_ATOMS * HDIM);
    gemm(xaggbf, pko[l], bo + (size_t)l * 3 * HDIM, obuf, N_ATOMS, 384, 128, 0);
    layer_update<<<(N_ATOMS * HDIM + 255) / 256, 256, 0, stream>>>(x, vec, obuf, vdot,
                                                                   vp, vagg);
  }

  // --- final layernorm + outputs: d_out = [x (N*H) | vec (N*3*H)] ---
  layernorm_k<0><<<N_ATOMS / 8, 256, 0, stream>>>(x, lnfg, lnfb, (float*)d_out);
  hipMemcpyAsync((float*)d_out + (size_t)N_ATOMS * HDIM, vec,
                 (size_t)N_ATOMS * 3 * HDIM * 4, hipMemcpyDeviceToDevice, stream);
}
